// RawSDP_34892314313293
// MI455X (gfx1250) — matrix operand order repaired and checked by reading, compiled, not run on hardware
//
#include <hip/hip_runtime.h>
#include <hip/hip_bf16.h>
#include <math.h>

typedef __attribute__((ext_vector_type(16))) _Float16 v16h;
typedef __attribute__((ext_vector_type(8)))  float    v8f;
typedef __attribute__((ext_vector_type(4)))  _Float16 h4;

namespace {
constexpr int B_   = 2;
constexpr int HQ   = 32;
constexpr int HKV  = 8;
constexpr int S_   = 2048;
constexpr int D_   = 128;
constexpr int TQ   = 128;   // q rows per block
constexpr int TKV  = 32;    // kv tile
constexpr int NW   = 8;     // waves per block
constexpr int KSTR = 136;   // K tile LDS stride (halves), 272B rows (16B aligned)
constexpr int VSTR = 40;    // V^T tile LDS stride (halves), 80B rows (16B aligned)
// log2(e) / sqrt(128): fold both the SDPA scale and the exp2 base change into Q
constexpr float QSCALE = 0.12751743f;
}

__global__ __launch_bounds__(256)
void fa_fwd(const float* __restrict__ Qg, const float* __restrict__ Kg,
            const float* __restrict__ Vg, float* __restrict__ Og)
{
  __shared__ _Float16 kT[TKV * KSTR];     // K tile, row-major [kv][d]
  __shared__ _Float16 vT[D_ * VSTR];      // V tile, transposed [d][kv]
  __shared__ _Float16 pT[NW * 16 * 32];   // per-wave P staging (C->A layout)

  const int tid  = threadIdx.x;
  const int lane = tid & 31;
  const int wave = tid >> 5;
  const int half = lane >> 4;   // which 16-lane half
  const int l15  = lane & 15;

  const int qt = blockIdx.x;
  const int h  = blockIdx.y;
  const int b  = blockIdx.z;
  const int hk = h / (HQ / HKV);

  const int qBlk = qt * TQ;
  const int qWav = qBlk + wave * 16;

  const float* Qp = Qg + (((size_t)b * HQ  + h ) * S_) * (size_t)D_;
  const float* Kp = Kg + (((size_t)b * HKV + hk) * S_) * (size_t)D_;
  const float* Vp = Vg + (((size_t)b * HKV + hk) * S_) * (size_t)D_;
  float*       Op = Og + (((size_t)b * HQ  + h ) * S_) * (size_t)D_;

  // ---- Q A-fragments: lane = M row; VGPRs 0-3 hold k 0..7 (+8 for upper half),
  // VGPRs 4-7 hold k 16..23 (+8 for upper half). Scale folded in. ----
  v16h qf[4];
  {
    const float* qrow = Qp + (size_t)(qWav + l15) * D_;
    #pragma unroll
    for (int kt = 0; kt < 4; ++kt) {
      const int dlo = kt * 32 + half * 8;
      const int dhi = kt * 32 + 16 + half * 8;
      #pragma unroll
      for (int i = 0; i < 8; ++i) {
        qf[kt][i]     = (_Float16)(qrow[dlo + i] * QSCALE);
        qf[kt][8 + i] = (_Float16)(qrow[dhi + i] * QSCALE);
      }
    }
  }

  v8f acc[8];
  #pragma unroll
  for (int dt = 0; dt < 8; ++dt) acc[dt] = {};
  float mrow[8], lrow[8];
  #pragma unroll
  for (int r = 0; r < 8; ++r) { mrow[r] = -3.0e38f; lrow[r] = 0.0f; }

  const int ntiles = (qBlk + TQ) / TKV;
  for (int t = 0; t < ntiles; ++t) {
    const int kv0 = t * TKV;

    // ---- cooperative K/V tile load, f32->f16, V transposed into LDS ----
    #pragma unroll
    for (int i = 0; i < 4; ++i) {
      const int idx = tid + i * 256;    // 0..1023 float4 slots
      const int row = idx >> 5;         // kv row 0..31
      const int c4  = idx & 31;         // d group of 4
      const float4 k4 = *(const float4*)(Kp + (size_t)(kv0 + row) * D_ + c4 * 4);
      h4 hk4;
      hk4[0] = (_Float16)k4.x; hk4[1] = (_Float16)k4.y;
      hk4[2] = (_Float16)k4.z; hk4[3] = (_Float16)k4.w;
      *(h4*)&kT[row * KSTR + c4 * 4] = hk4;
      const float4 v4 = *(const float4*)(Vp + (size_t)(kv0 + row) * D_ + c4 * 4);
      vT[(c4 * 4 + 0) * VSTR + row] = (_Float16)v4.x;
      vT[(c4 * 4 + 1) * VSTR + row] = (_Float16)v4.y;
      vT[(c4 * 4 + 2) * VSTR + row] = (_Float16)v4.z;
      vT[(c4 * 4 + 3) * VSTR + row] = (_Float16)v4.w;
    }
    __syncthreads();

    if (kv0 <= qWav + 15) {   // wave-uniform: EXEC stays all-1s for WMMA
      // ---- S = Q K^T : two 16x16 N-tiles over this kv tile ----
      v8f sc[2];
      #pragma unroll
      for (int nt = 0; nt < 2; ++nt) {
        v8f c = {};
        #pragma unroll
        for (int kt = 0; kt < 4; ++kt) {
          // B frag: lane = kv column; d 0..7 and 16..23 (upper lanes +8)
          v16h bf;
          const _Float16* src = &kT[(l15 + 16 * nt) * KSTR + kt * 32 + half * 8];
          #pragma unroll
          for (int i = 0; i < 16; ++i) bf[i] = src[i + (i >= 8 ? 8 : 0)];
          c = __builtin_amdgcn_wmma_f32_16x16x32_f16(false, qf[kt], false, bf,
                                                     (short)0, c, false, false);
        }
        sc[nt] = c;
      }

      // ---- causal mask + online softmax (C layout: VGPR r -> rows r / r+8) ----
      const bool needMask = (kv0 + TKV - 1) > qWav;
      float mnew[8];
      #pragma unroll
      for (int r = 0; r < 8; ++r) {
        const int rowg = qWav + r + 8 * half;
        if (needMask) {
          if (kv0 + l15 > rowg)      sc[0][r] = -3.0e38f;
          if (kv0 + 16 + l15 > rowg) sc[1][r] = -3.0e38f;
        }
        float v = fmaxf((float)sc[0][r], (float)sc[1][r]);
        v = fmaxf(v, __shfl_xor(v, 1));
        v = fmaxf(v, __shfl_xor(v, 2));
        v = fmaxf(v, __shfl_xor(v, 4));
        v = fmaxf(v, __shfl_xor(v, 8));
        mnew[r] = fmaxf(mrow[r], v);
      }
      #pragma unroll
      for (int r = 0; r < 8; ++r) {
        const float p0 = exp2f((float)sc[0][r] - mnew[r]);
        const float p1 = exp2f((float)sc[1][r] - mnew[r]);
        sc[0][r] = p0; sc[1][r] = p1;
        float s = p0 + p1;
        s += __shfl_xor(s, 1);
        s += __shfl_xor(s, 2);
        s += __shfl_xor(s, 4);
        s += __shfl_xor(s, 8);
        const float alpha = exp2f(mrow[r] - mnew[r]);
        lrow[r] = lrow[r] * alpha + s;
        mrow[r] = mnew[r];
        #pragma unroll
        for (int dt = 0; dt < 8; ++dt) acc[dt][r] = acc[dt][r] * alpha;
      }

      // ---- P: C-layout regs -> A-layout via per-wave LDS staging ----
      _Float16* pw = &pT[wave * 512];
      #pragma unroll
      for (int r = 0; r < 8; ++r) {
        const int row = r + 8 * half;
        pw[row * 32 + l15]      = (_Float16)sc[0][r];
        pw[row * 32 + 16 + l15] = (_Float16)sc[1][r];
      }
      asm volatile("s_wait_dscnt 0" ::: "memory");  // DS in-order per wave
      v16h pf;
      {
        const _Float16* p0 = &pw[l15 * 32 + half * 8];
        const _Float16* p1 = &pw[l15 * 32 + 16 + half * 8];
        #pragma unroll
        for (int i = 0; i < 8; ++i) { pf[i] = p0[i]; pf[8 + i] = p1[i]; }
      }

      // ---- O += P V : 8 d-tiles ----
      #pragma unroll
      for (int dt = 0; dt < 8; ++dt) {
        v16h vf;  // lane = d column; kv 0..7 and 16..23 (upper lanes +8)
        const _Float16* vs = &vT[(dt * 16 + l15) * VSTR + half * 8];
        #pragma unroll
        for (int i = 0; i < 16; ++i) vf[i] = vs[i + (i >= 8 ? 8 : 0)];
        acc[dt] = __builtin_amdgcn_wmma_f32_16x16x32_f16(false, pf, false, vf,
                                                         (short)0, acc[dt], false, false);
      }
    }
    __syncthreads();
  }

  // ---- epilogue: O = acc / l ----
  #pragma unroll
  for (int r = 0; r < 8; ++r) {
    const float inv = 1.0f / lrow[r];
    const size_t row = (size_t)(qWav + r + 8 * half);
    #pragma unroll
    for (int dt = 0; dt < 8; ++dt) {
      Op[row * D_ + dt * 16 + l15] = acc[dt][r] * inv;
    }
  }
}

extern "C" void kernel_launch(void* const* d_in, const int* in_sizes, int n_in,
                              void* d_out, int out_size, void* d_ws, size_t ws_size,
                              hipStream_t stream) {
  const float* Q = (const float*)d_in[0];
  const float* K = (const float*)d_in[1];
  const float* V = (const float*)d_in[2];
  float*       O = (float*)d_out;
  dim3 grid(S_ / TQ, HQ, B_);
  fa_fwd<<<grid, dim3(256), 0, stream>>>(Q, K, V, O);
}
